// ResidualGNNsWithEdgeLevelAttention_66975720013952
// MI455X (gfx1250) — matrix operand order repaired and checked by reading, not compiled, not run on hardware
//
#include <hip/hip_runtime.h>
#include <hip/hip_bf16.h>

// ---------------------------------------------------------------------------
// Problem constants (from the reference)
// ---------------------------------------------------------------------------
#define NNODES 16384
#define FDIM   256
#define NG     64          // graphs
#define NEDGE  131072
#define NLAY   3
#define HID    512
#define IU     32640       // F*(F-1)/2
#define IN1    33408       // IU + F*L
#define EPSBN  1e-5f

typedef __attribute__((ext_vector_type(16))) __bf16 v16bf;
typedef __attribute__((ext_vector_type(8)))  float  v8f;
typedef unsigned int u32x4 __attribute__((ext_vector_type(4)));
typedef int          i32x8 __attribute__((ext_vector_type(8)));
typedef int          i32x4 __attribute__((ext_vector_type(4)));

// ---------------------------------------------------------------------------
// The TDM writes LDS through a descriptor the compiler cannot alias-analyze.
// Escaping the LDS pointer into opaque asm with a memory clobber makes the
// compiler assume the array contents are defined (round 2: loads folded to
// undef without this). Loads stay NON-volatile so InferAddressSpaces can
// rewrite them to ds_load (round 3: volatile loads stayed generic -> flat).
// ---------------------------------------------------------------------------
__device__ __forceinline__ void lds_publish(const void* p) {
    asm volatile("" : : "r"(p) : "memory");
}

// ---------------------------------------------------------------------------
// Tensor Data Mover: 2-D fp32 tile (global -> LDS), D# per ISA 08 §8.
//   lds_off      : byte offset in the workgroup's LDS
//   gaddr        : global address of the TILE start
//   tensor_d0/d1 : tensor extents (elements) for OOB checks
//   tile_d0/d1   : tile extents (elements); d0 is the contiguous dim
//   d0_stride    : elements between consecutive rows
// ---------------------------------------------------------------------------
__device__ __forceinline__ void tdm_load_2d_f32(unsigned lds_off, const void* gaddr,
                                                unsigned tensor_d0, unsigned tensor_d1,
                                                unsigned tile_d0, unsigned tile_d1,
                                                unsigned d0_stride) {
    unsigned long long ga = (unsigned long long)gaddr;
    u32x4 g0;
    g0[0] = 1u;                                            // count=1, user D#
    g0[1] = lds_off;                                       // lds_addr
    g0[2] = (unsigned)(ga & 0xFFFFFFFFu);                  // global_addr lo
    g0[3] = (unsigned)((ga >> 32) & 0x01FFFFFFu) | (2u << 30); // addr hi | type=2
    i32x8 g1;
    g1[0] = (int)(2u << 16);                               // data_size=4B, no mcast
    g1[1] = (int)((tensor_d0 & 0xFFFFu) << 16);            // tensor_dim0 lo
    g1[2] = (int)(((tensor_d0 >> 16) & 0xFFFFu) |
                  ((tensor_d1 & 0xFFFFu) << 16));          // dim0 hi | dim1 lo
    g1[3] = (int)(((tensor_d1 >> 16) & 0xFFFFu) |
                  (tile_d0 << 16));                        // dim1 hi | tile_dim0
    g1[4] = (int)(tile_d1 & 0xFFFFu);                      // tile_dim1, tile_dim2=0
    g1[5] = (int)d0_stride;                                // tensor_dim0_stride lo
    g1[6] = 0;                                             // stride hi, dim1_stride
    g1[7] = 0;
    i32x4 z4 = {0, 0, 0, 0};
#if __clang_major__ >= 23
    i32x8 z8 = {0, 0, 0, 0, 0, 0, 0, 0};
    __builtin_amdgcn_tensor_load_to_lds(g0, g1, z4, z4, z8, 0);
#else
    __builtin_amdgcn_tensor_load_to_lds(g0, g1, z4, z4, 0);
#endif
}

// ---------------------------------------------------------------------------
// Init: deg=1 (self loop), pooled sums = 0, out1raw seeded with bias b1
// ---------------------------------------------------------------------------
__global__ void k_init(float* degf, float* pooled, float* out1raw,
                       const float* __restrict__ b1) {
    int t = blockIdx.x * blockDim.x + threadIdx.x;
    if (t < NNODES)           degf[t]    = 1.0f;
    if (t < NLAY * NG * FDIM) pooled[t]  = 0.0f;
    if (t < NG * HID)         out1raw[t] = b1[t & (HID - 1)];
}

// ---------------------------------------------------------------------------
// Degree histogram over edge destinations
// ---------------------------------------------------------------------------
__global__ void k_deg(const int* __restrict__ dst, float* degf) {
    int e = blockIdx.x * blockDim.x + threadIdx.x;
    if (e < NEDGE) atomicAdd(&degf[dst[e]], 1.0f);
}

// ---------------------------------------------------------------------------
// Conv layer (attention is algebraically the identity):
//   hout = diag(deg) * (hin @ W + bias),   W: [256,256] row-major (k,n)
// Weight tile (256 x 64 fp32, 64 KB) staged in LDS by the Tensor Data Mover.
// Each wave computes a 32x32 tile (2x2 WMMA accs), 8 waves -> 128x64 per WG.
// grid = (16384/128, 256/64) = (128, 4), block = 256.
// ---------------------------------------------------------------------------
__global__ void __launch_bounds__(256)
k_conv(const float* __restrict__ hin, const float* __restrict__ W,
       const float* __restrict__ bias, const float* __restrict__ degf,
       float* __restrict__ hout, float* __restrict__ pooled) {
    __shared__ float ldsW[FDIM * 64];          // sole static LDS -> offset 0

    const int wave  = threadIdx.x >> 5;
    const int lane  = threadIdx.x & 31;
    const int laneL = lane & 15;     // M for A-frag, N for B/C/D frags
    const int kHalf = lane >> 4;     // which K-half this lane holds
    const int rowBase = blockIdx.x * 128 + (wave & 3) * 32;
    const int lcolBase = (wave >> 2) * 32;           // column within LDS tile
    const int colBase  = blockIdx.y * 64 + lcolBase; // global column

    // --- TDM: one DMA pulls the whole 256x64 weight tile into LDS ----------
    if (threadIdx.x < 32) {
        tdm_load_2d_f32(/*lds_off=*/0u, W + blockIdx.y * 64,
                        /*tensor_d0=*/FDIM, /*tensor_d1=*/FDIM,
                        /*tile_d0=*/64, /*tile_d1=*/FDIM,
                        /*d0_stride=*/FDIM);
        __builtin_amdgcn_s_wait_tensorcnt(0);
    }
    __syncthreads();
    lds_publish(ldsW);               // LDS now holds TDM-written data

    v8f acc[2][2] = {};

    for (int kb = 0; kb < FDIM; kb += 32) {
        // ---- A fragments: 16-bit A 16x32 layout (ISA 7.12.2) ----
        v16bf a[2];
#pragma unroll
        for (int mi = 0; mi < 2; ++mi) {
            const float* p = hin + (rowBase + mi * 16 + laneL) * FDIM
                                 + kb + kHalf * 8;
            __builtin_prefetch(p + 32, 0, 1);           // next K tile
            float4 f0 = *(const float4*)(p);
            float4 f1 = *(const float4*)(p + 4);
            float4 f2 = *(const float4*)(p + 16);
            float4 f3 = *(const float4*)(p + 20);
            a[mi][0]  = (__bf16)f0.x; a[mi][1]  = (__bf16)f0.y;
            a[mi][2]  = (__bf16)f0.z; a[mi][3]  = (__bf16)f0.w;
            a[mi][4]  = (__bf16)f1.x; a[mi][5]  = (__bf16)f1.y;
            a[mi][6]  = (__bf16)f1.z; a[mi][7]  = (__bf16)f1.w;
            a[mi][8]  = (__bf16)f2.x; a[mi][9]  = (__bf16)f2.y;
            a[mi][10] = (__bf16)f2.z; a[mi][11] = (__bf16)f2.w;
            a[mi][12] = (__bf16)f3.x; a[mi][13] = (__bf16)f3.y;
            a[mi][14] = (__bf16)f3.z; a[mi][15] = (__bf16)f3.w;
        }
        // ---- B fragments from LDS: 16-bit B 32x16, table K order (matches A) --
        v16bf bm[2];
#pragma unroll
        for (int ni = 0; ni < 2; ++ni) {
            const float* q = ldsW + (kb + kHalf * 8) * 64
                                  + lcolBase + ni * 16 + laneL;
#pragma unroll
            for (int i = 0; i < 16; ++i) bm[ni][i] = (__bf16)q[(i + (i >> 3) * 8) * 64];
        }
#pragma unroll
        for (int mi = 0; mi < 2; ++mi)
#pragma unroll
            for (int ni = 0; ni < 2; ++ni)
                acc[mi][ni] = __builtin_amdgcn_wmma_f32_16x16x32_bf16(
                    false, a[mi], false, bm[ni], (short)0, acc[mi][ni],
                    false, false);
    }

    // Epilogue: bias, degree scale, store, fused column-sum pooling
#pragma unroll
    for (int mi = 0; mi < 2; ++mi) {
        const int mTile = rowBase + mi * 16;
        const int gidx  = mTile >> 8;              // tile lies in one graph
#pragma unroll
        for (int ni = 0; ni < 2; ++ni) {
            const int n  = colBase + ni * 16 + laneL;
            const float bn = bias[n];
            float colsum = 0.f;
#pragma unroll
            for (int r = 0; r < 8; ++r) {
                const int m = mTile + r + kHalf * 8;  // C/D layout
                const float v = (acc[mi][ni][r] + bn) * degf[m];
                hout[m * FDIM + n] = v;
                colsum += v;
            }
            colsum += __shfl_xor(colsum, 16, 32);
            if (lane < 16) atomicAdd(&pooled[gidx * FDIM + n], colsum);
        }
    }
}

// ---------------------------------------------------------------------------
// Strict-upper-triangle extraction + BatchNorm over G=64 -> z[:, 0:IU]
// grid = 255 blocks (row i), 256 threads (col j = i+1+tid)
// ---------------------------------------------------------------------------
__global__ void k_xt_bn(const float* __restrict__ x, const float* __restrict__ g,
                        const float* __restrict__ b, float* __restrict__ z) {
    const int i = blockIdx.x;
    const int j = i + 1 + threadIdx.x;
    if (j >= FDIM) return;
    const int c = (i * (2 * FDIM - i - 1)) / 2 + (j - i - 1);
    float s = 0.f, ss = 0.f;
    for (int gi = 0; gi < NG; ++gi) {
        float v = x[((gi * FDIM + i) << 8) + j];
        s += v; ss += v * v;
    }
    const float m   = s * (1.f / NG);
    const float var = ss * (1.f / NG) - m * m;
    const float inv = rsqrtf(var + EPSBN);
    const float gg = g[c], bb = b[c];
    for (int gi = 0; gi < NG; ++gi) {
        float v = x[((gi * FDIM + i) << 8) + j];
        z[gi * IN1 + c] = (v - m) * inv * gg + bb;
    }
}

// ---------------------------------------------------------------------------
// Pooled means + BatchNorm -> z[:, IU:IU+768]
// ---------------------------------------------------------------------------
__global__ void k_hh_bn(const float* __restrict__ pooled,
                        const float* __restrict__ g, const float* __restrict__ b,
                        float* __restrict__ z) {
    const int c = blockIdx.x * blockDim.x + threadIdx.x;
    if (c >= NLAY * FDIM) return;
    const int l = c >> 8, f = c & 255;
    const float* p = pooled + l * NG * FDIM;
    float s = 0.f, ss = 0.f;
    for (int gi = 0; gi < NG; ++gi) {
        float v = p[gi * FDIM + f] * (1.f / FDIM);   // mean pool (cnt = 256)
        s += v; ss += v * v;
    }
    const float m   = s * (1.f / NG);
    const float var = ss * (1.f / NG) - m * m;
    const float inv = rsqrtf(var + EPSBN);
    const float gg = g[c], bb = b[c];
    for (int gi = 0; gi < NG; ++gi) {
        float v = p[gi * FDIM + f] * (1.f / FDIM);
        z[gi * IN1 + IU + c] = (v - m) * inv * gg + bb;
    }
}

// ---------------------------------------------------------------------------
// MLP layer 1: [64,33408] @ [33408,512], split-K bf16 WMMA, atomic reduce.
// w1 K-chunks (32x16 fp32, 2 KB) streamed through a double-buffered LDS
// pipeline driven by the Tensor Data Mover (issue k+1, wait tensorcnt<=1,
// consume k). grid = (4*32 tiles, 12 K-chunks of 87 iters), block = 32.
// ---------------------------------------------------------------------------
#define K1ITER 87   // 33408 / 32 = 1044 = 12 * 87
__global__ void __launch_bounds__(32)
k_mlp1(const float* __restrict__ z, const float* __restrict__ w1,
       float* __restrict__ out) {
    __shared__ float ldsB[2 * 32 * 16];        // sole static LDS -> offset 0

    const int lane  = threadIdx.x & 31;
    const int laneL = lane & 15;
    const int kHalf = lane >> 4;
    const int tM = (blockIdx.x & 3) * 16;
    const int tN = (blockIdx.x >> 2) * 16;
    const int k0 = blockIdx.y * (K1ITER * 32);

    // Prime the pipeline: chunk 0 -> buffer 0
    tdm_load_2d_f32(0u, w1 + (unsigned long long)k0 * HID + tN,
                    /*tensor_d0=*/HID, /*tensor_d1=*/IN1,
                    /*tile_d0=*/16, /*tile_d1=*/32, /*d0_stride=*/HID);

    v8f acc = {};
    for (int kk = 0; kk < K1ITER; ++kk) {
        if (kk + 1 < K1ITER) {
            const int kn = k0 + (kk + 1) * 32;
            tdm_load_2d_f32((unsigned)(((kk + 1) & 1) * 2048),
                            w1 + (unsigned long long)kn * HID + tN,
                            HID, IN1, 16, 32, HID);
            __builtin_amdgcn_s_wait_tensorcnt(1);   // chunk kk has landed
        } else {
            __builtin_amdgcn_s_wait_tensorcnt(0);   // last chunk landed
        }
        lds_publish(ldsB);                          // TDM data now visible

        const int kb = k0 + kk * 32;
        v16bf a, bm;
        const float* p = z + (tM + laneL) * IN1 + kb + kHalf * 8;
        float4 f0 = *(const float4*)(p);
        float4 f1 = *(const float4*)(p + 4);
        float4 f2 = *(const float4*)(p + 16);
        float4 f3 = *(const float4*)(p + 20);
        a[0]  = (__bf16)f0.x; a[1]  = (__bf16)f0.y;
        a[2]  = (__bf16)f0.z; a[3]  = (__bf16)f0.w;
        a[4]  = (__bf16)f1.x; a[5]  = (__bf16)f1.y;
        a[6]  = (__bf16)f1.z; a[7]  = (__bf16)f1.w;
        a[8]  = (__bf16)f2.x; a[9]  = (__bf16)f2.y;
        a[10] = (__bf16)f2.z; a[11] = (__bf16)f2.w;
        a[12] = (__bf16)f3.x; a[13] = (__bf16)f3.y;
        a[14] = (__bf16)f3.z; a[15] = (__bf16)f3.w;

        const float* q = ldsB + (kk & 1) * 512 + (kHalf * 8) * 16 + laneL;
#pragma unroll
        for (int i = 0; i < 16; ++i) bm[i] = (__bf16)q[(i + (i >> 3) * 8) * 16];

        acc = __builtin_amdgcn_wmma_f32_16x16x32_bf16(
            false, a, false, bm, (short)0, acc, false, false);
    }
    const int n = tN + laneL;
#pragma unroll
    for (int r = 0; r < 8; ++r) {
        const int m = tM + r + kHalf * 8;
        atomicAdd(&out[m * HID + n], acc[r]);
    }
}

// ---------------------------------------------------------------------------
// Per-column BatchNorm (biased var) + ReLU over the 64-row batch
// ---------------------------------------------------------------------------
__global__ void k_bn_relu(const float* __restrict__ in, const float* __restrict__ g,
                          const float* __restrict__ b, float* __restrict__ out,
                          int C) {
    const int c = blockIdx.x * blockDim.x + threadIdx.x;
    if (c >= C) return;
    float s = 0.f, ss = 0.f;
    for (int gi = 0; gi < NG; ++gi) {
        float v = in[gi * C + c];
        s += v; ss += v * v;
    }
    const float m   = s * (1.f / NG);
    const float var = ss * (1.f / NG) - m * m;
    const float inv = rsqrtf(var + EPSBN);
    const float gg = g[c], bb = b[c];
    for (int gi = 0; gi < NG; ++gi) {
        float v = (in[gi * C + c] - m) * inv * gg + bb;
        out[gi * C + c] = v > 0.f ? v : 0.f;
    }
}

// ---------------------------------------------------------------------------
// Tiny FC layers on the 64-row batch (negligible flops -> plain VALU)
// ---------------------------------------------------------------------------
__global__ void k_fc(const float* __restrict__ in, const float* __restrict__ W,
                     const float* __restrict__ bias, float* __restrict__ out,
                     int K, int Cn) {
    const int t = blockIdx.x * blockDim.x + threadIdx.x;
    if (t >= NG * Cn) return;
    const int m = t / Cn, n = t % Cn;
    float acc = bias[n];
    for (int k = 0; k < K; ++k) acc += in[m * K + k] * W[k * Cn + n];
    out[t] = acc;
}

// ---------------------------------------------------------------------------
// Host-side launch
// ---------------------------------------------------------------------------
extern "C" void kernel_launch(void* const* d_in, const int* in_sizes, int n_in,
                              void* d_out, int out_size, void* d_ws, size_t ws_size,
                              hipStream_t stream) {
    const float* x         = (const float*)d_in[0];
    const int*   edge      = (const int*)  d_in[1];
    const float* lin_dst_w = (const float*)d_in[5];
    const float* lin_dst_b = (const float*)d_in[6];
    const float* bn_g  = (const float*)d_in[9];
    const float* bn_b  = (const float*)d_in[10];
    const float* bnh_g = (const float*)d_in[11];
    const float* bnh_b = (const float*)d_in[12];
    const float* w1 = (const float*)d_in[13];
    const float* b1 = (const float*)d_in[14];
    const float* g1 = (const float*)d_in[15];
    const float* be1= (const float*)d_in[16];
    const float* w2 = (const float*)d_in[17];
    const float* b2 = (const float*)d_in[18];
    const float* g2 = (const float*)d_in[19];
    const float* be2= (const float*)d_in[20];
    const float* w3 = (const float*)d_in[21];
    const float* b3 = (const float*)d_in[22];
    const float* g3 = (const float*)d_in[23];
    const float* be3= (const float*)d_in[24];
    const float* w4 = (const float*)d_in[25];
    const float* b4 = (const float*)d_in[26];
    float* out = (float*)d_out;

    // Workspace layout (all 256B-aligned sizes), ~43 MB total.
    char* w = (char*)d_ws;
    float* degf    = (float*)w;  w += NNODES * 4;                 // 64 KB
    float* pooled  = (float*)w;  w += NLAY * NG * FDIM * 4;       // 192 KB
    float* hA      = (float*)w;  w += NNODES * FDIM * 4;          // 16 MB
    float* hB      = (float*)w;  w += NNODES * FDIM * 4;          // 16 MB
    float* z       = (float*)w;  w += NG * IN1 * 4;               // 8.2 MB
    float* out1raw = (float*)w;  w += NG * HID * 4;
    float* z1      = (float*)w;  w += NG * HID * 4;
    float* out2raw = (float*)w;  w += NG * (HID/2) * 4;
    float* z2      = (float*)w;  w += NG * (HID/2) * 4;
    float* out3raw = (float*)w;  w += NG * (HID/2) * 4;
    float* z3      = (float*)w;  w += NG * (HID/2) * 4;

    // 1) init scratch state
    k_init<<<(NLAY * NG * FDIM + 255) / 256, 256, 0, stream>>>(degf, pooled,
                                                               out1raw, b1);
    // 2) degree histogram (dst = edge_index[1] = edge + E)
    k_deg<<<NEDGE / 256, 256, 0, stream>>>(edge + NEDGE, degf);

    // 3) three conv layers (ping-pong h buffers), pooling fused
    dim3 cgrid(NNODES / 128, FDIM / 64);
    k_conv<<<cgrid, 256, 0, stream>>>(x,  lin_dst_w + 0 * FDIM * FDIM,
                                      lin_dst_b + 0 * FDIM, degf, hA,
                                      pooled + 0 * NG * FDIM);
    k_conv<<<cgrid, 256, 0, stream>>>(hA, lin_dst_w + 1 * FDIM * FDIM,
                                      lin_dst_b + 1 * FDIM, degf, hB,
                                      pooled + 1 * NG * FDIM);
    k_conv<<<cgrid, 256, 0, stream>>>(hB, lin_dst_w + 2 * FDIM * FDIM,
                                      lin_dst_b + 2 * FDIM, degf, hA,
                                      pooled + 2 * NG * FDIM);

    // 4) build MLP input z = [BN(triu(x)) | BN(pooled means)]
    k_xt_bn<<<FDIM - 1, 256, 0, stream>>>(x, bn_g, bn_b, z);
    k_hh_bn<<<3, 256, 0, stream>>>(pooled, bnh_g, bnh_b, z);

    // 5) MLP layer 1 (bandwidth-bound on w1): TDM-pipelined split-K WMMA
    k_mlp1<<<dim3(128, 12), 32, 0, stream>>>(z, w1, out1raw);
    k_bn_relu<<<HID / 256, 256, 0, stream>>>(out1raw, g1, be1, z1, HID);

    // 6) tail layers (tiny)
    k_fc<<<(NG * (HID/2) + 255) / 256, 256, 0, stream>>>(z1, w2, b2, out2raw,
                                                         HID, HID/2);
    k_bn_relu<<<1, 256, 0, stream>>>(out2raw, g2, be2, z2, HID/2);
    k_fc<<<(NG * (HID/2) + 255) / 256, 256, 0, stream>>>(z2, w3, b3, out3raw,
                                                         HID/2, HID/2);
    k_bn_relu<<<1, 256, 0, stream>>>(out3raw, g3, be3, z3, HID/2);
    k_fc<<<1, 128, 0, stream>>>(z3, w4, b4, out, HID/2, 2);
}
